// MS_MSA_69372311765381
// MI455X (gfx1250) — matrix operand order repaired and checked by reading, not compiled, not run on hardware
//
#include <hip/hip_runtime.h>
#include <hip/hip_bf16.h>
#include <math.h>

// ---------------------------------------------------------------------------
// CDNA5 (gfx1250) WMMA helpers: wave32, v_wmma_f32_16x16x32_bf16
// ---------------------------------------------------------------------------
typedef __attribute__((ext_vector_type(16))) __bf16 v16bf;
typedef __attribute__((ext_vector_type(8)))  float  v8f;

union FragU { v16bf v; uint4 q[2]; };

__device__ __forceinline__ unsigned short f2bf(float f) {
    unsigned u = __float_as_uint(f);
    u += 0x7fffu + ((u >> 16) & 1u);          // round-to-nearest-even
    return (unsigned short)(u >> 16);
}
__device__ __forceinline__ float bf2f(unsigned short h) {
    return __uint_as_float(((unsigned)h) << 16);
}

// A fragment: 16x32 bf16, row-major source [M][K], ld in elements.
// Lane L<16  -> row M=L,  K in {0..7, 16..23}
// Lane L>=16 -> row M=L-16, K in {8..15, 24..31}   (ISA 7.12.2)
__device__ __forceinline__ v16bf load_a_frag(const unsigned short* base, int ld) {
    int lane = threadIdx.x & 31;
    int r = lane & 15, hi = lane >> 4;
    const unsigned short* rowp = base + (size_t)r * ld;
    FragU u;
    u.q[0] = *(const uint4*)(rowp + (hi ? 8 : 0));
    u.q[1] = *(const uint4*)(rowp + (hi ? 24 : 16));
    return u.v;
}

// B fragment: 32x16 bf16 from transposed source Bt[N][K] row-major.
// Lane L<16 -> col N=L, K in {0..7, 16..23} ; Lane L>=16 -> col N=L-16, K in {8..15, 24..31}.
__device__ __forceinline__ v16bf load_b_frag(const unsigned short* base, int ld) {
    int lane = threadIdx.x & 31;
    int n = lane & 15, hi = lane >> 4;
    const unsigned short* rowp = base + (size_t)n * ld + hi * 8;
    FragU u;
    u.q[0] = *(const uint4*)(rowp);
    u.q[1] = *(const uint4*)(rowp + 16);
    return u.v;
}

__device__ __forceinline__ v8f wmma_bf16(v16bf a, v16bf b, v8f c) {
    return __builtin_amdgcn_wmma_f32_16x16x32_bf16(false, a, false, b,
                                                   (short)0, c, false, false);
}

// ---------------------------------------------------------------------------
// Problem constants
// ---------------------------------------------------------------------------
#define BB    2
#define HH    256
#define WW    256
#define CC    64
#define HEADS 8
#define DH    64
#define HD    512
#define NTOK  65536     // H*W
#define NDS   16384     // (H/2)*(W/2)

// ---------------------------------------------------------------------------
// K1: avgpool 2x2 -> xd (bf16), and cast full-res x -> bf16
// ---------------------------------------------------------------------------
__global__ void k_pool_cast(const float* __restrict__ x,
                            unsigned short* __restrict__ x_bf,
                            unsigned short* __restrict__ xd_bf) {
    int idx = blockIdx.x * 256 + threadIdx.x;       // B*128*128*64 = 2M
    int c = idx & 63;
    int j = (idx >> 6) & 127;
    int i = (idx >> 13) & 127;
    int b = idx >> 20;
    size_t base = ((((size_t)b * HH + 2 * i) * WW) + 2 * j) * CC + c;
    float v00 = x[base];
    float v01 = x[base + CC];
    float v10 = x[base + (size_t)WW * CC];
    float v11 = x[base + (size_t)WW * CC + CC];
    x_bf[base] = f2bf(v00);
    x_bf[base + CC] = f2bf(v01);
    x_bf[base + (size_t)WW * CC] = f2bf(v10);
    x_bf[base + (size_t)WW * CC + CC] = f2bf(v11);
    xd_bf[(((size_t)b * NDS) + i * 128 + j) * CC + c] =
        f2bf(0.25f * (v00 + v01 + v10 + v11));
}

// ---------------------------------------------------------------------------
// K2: pack [Wq | Wk] transposed to bf16  ->  Wqkt[1024][64]
// ---------------------------------------------------------------------------
__global__ void k_prep_wqk(const float* __restrict__ Wq,
                           const float* __restrict__ Wk,
                           unsigned short* __restrict__ Wqkt) {
    int idx = blockIdx.x * 256 + threadIdx.x;       // 1024*64
    int k = idx & 63;
    int n = idx >> 6;
    float v = (n < HD) ? Wq[k * HD + n] : Wk[k * HD + (n - HD)];
    Wqkt[n * 64 + k] = f2bf(v);
}

// ---------------------------------------------------------------------------
// K3: Q/K projection GEMM (M=32768, N=1024, K=64) with WMMA bf16.
// Output stored TRANSPOSED: Qt/Kt [b][512][16384] bf16 (token-contiguous).
// ---------------------------------------------------------------------------
__global__ void k_gemm_qk(const unsigned short* __restrict__ xd,
                          const unsigned short* __restrict__ Wqkt,
                          unsigned short* __restrict__ Qt,
                          unsigned short* __restrict__ Kt) {
    int wave = threadIdx.x >> 5;
    int lane = threadIdx.x & 31;
    int m0 = blockIdx.x * 16;                       // token tile base
    int ntile = blockIdx.y * 8 + wave;              // 0..63

    const unsigned short* arow = xd + (size_t)m0 * 64;
    v16bf a0 = load_a_frag(arow, 64);               // K 0..31
    v16bf a1 = load_a_frag(arow + 32, 64);          // K 32..63
    const unsigned short* brow = Wqkt + (size_t)ntile * 16 * 64;
    v16bf b0 = load_b_frag(brow, 64);
    v16bf b1 = load_b_frag(brow + 32, 64);

    v8f acc = {};
    acc = wmma_bf16(a0, b0, acc);
    acc = wmma_bf16(a1, b1, acc);

    int n = ntile * 16 + (lane & 15);               // global out channel 0..1023
    int hi = lane >> 4;
    int b = m0 >> 14;                               // 16384 tokens per batch
    int tloc = (m0 & (NDS - 1)) + hi * 8;
    unsigned short* dst = (n < HD) ? Qt : Kt;
    int nn = (n < HD) ? n : n - HD;
    unsigned short h8[8];
#pragma unroll
    for (int r = 0; r < 8; ++r) h8[r] = f2bf(acc[r]);
    *(uint4*)(dst + ((size_t)b * HD + nn) * NDS + tloc) = *(uint4*)h8;
}

// ---------------------------------------------------------------------------
// K4: per-channel inverse L2 norms over tokens: invq/invk [b*512]
// ---------------------------------------------------------------------------
__global__ void k_norms(const unsigned short* __restrict__ Qt,
                        const unsigned short* __restrict__ Kt,
                        float* __restrict__ invq, float* __restrict__ invk) {
    int wave = threadIdx.x >> 5, lane = threadIdx.x & 31;
    int row = blockIdx.x * 8 + wave;                // 0..1023
    const unsigned short* src =
        (blockIdx.y == 0 ? Qt : Kt) + (size_t)row * NDS;
    float s = 0.f;
    for (int it = 0; it < 64; ++it) {
        uint4 q = *(const uint4*)(src + it * 256 + lane * 8);
        unsigned short h[8];
        *(uint4*)h = q;
#pragma unroll
        for (int e = 0; e < 8; ++e) { float f = bf2f(h[e]); s += f * f; }
    }
#pragma unroll
    for (int off = 16; off >= 1; off >>= 1) s += __shfl_down(s, off, 32);
    if (lane == 0) {
        float inv = 1.0f / fmaxf(sqrtf(s), 1e-12f);
        (blockIdx.y == 0 ? invq : invk)[row] = inv;
    }
}

// ---------------------------------------------------------------------------
// K5: per-(b,h) channel attention: attn = softmax(norm(K)^T norm(Q) * rescale)
// via WMMA over Kdim=16384 tokens; then fold: Wmix[h*64+j][o] =
// sum_i attn[i][j] * Wproj[h*64+i][o].
// One block per (b,h); 8 waves own 2 of the 16 16x16 tiles each.
// ---------------------------------------------------------------------------
__global__ void k_attn(const unsigned short* __restrict__ Qt,
                       const unsigned short* __restrict__ Kt,
                       const float* __restrict__ invq,
                       const float* __restrict__ invk,
                       const float* __restrict__ rescale,
                       const float* __restrict__ Wproj,
                       float* __restrict__ Wmix) {
    __shared__ float attn[64][65];
    int bh = blockIdx.x, b = bh >> 3, h = bh & 7;
    int wave = threadIdx.x >> 5, lane = threadIdx.x & 31;
    int t0 = wave * 2;
    int mi = t0 >> 2;                               // K-channel tile (i)
    int nj0 = t0 & 3;                               // Q-channel tile (j)

    const unsigned short* Kbase = Kt + ((size_t)b * HD + h * DH + mi * 16) * NDS;
    const unsigned short* Qb0  = Qt + ((size_t)b * HD + h * DH + nj0 * 16) * NDS;
    const unsigned short* Qb1  = Qb0 + (size_t)16 * NDS;

    v8f acc0 = {}, acc1 = {};
    for (int tk = 0; tk < NDS; tk += 32) {
        v16bf a  = load_a_frag(Kbase + tk, NDS);    // A = K^T (channel rows)
        v16bf bb0 = load_b_frag(Qb0 + tk, NDS);     // B = Q   (channel cols)
        v16bf bb1 = load_b_frag(Qb1 + tk, NDS);
        acc0 = wmma_bf16(a, bb0, acc0);
        acc1 = wmma_bf16(a, bb1, acc1);
    }

    int ib = mi * 16 + (lane >> 4) * 8;
    int jn0 = nj0 * 16 + (lane & 15);
    int jn1 = jn0 + 16;
    float rs = rescale[h];
    float iq0 = invq[b * HD + h * DH + jn0] * rs;
    float iq1 = invq[b * HD + h * DH + jn1] * rs;
#pragma unroll
    for (int r = 0; r < 8; ++r) {
        int i = ib + r;
        float ik = invk[b * HD + h * DH + i];
        attn[i][jn0] = acc0[r] * ik * iq0;
        attn[i][jn1] = acc1[r] * ik * iq1;
    }
    __syncthreads();

    if (threadIdx.x < 64) {                         // softmax over j per row i
        int i = threadIdx.x;
        float m = -3.4e38f;
        for (int j = 0; j < 64; ++j) m = fmaxf(m, attn[i][j]);
        float sum = 0.f;
        for (int j = 0; j < 64; ++j) { float e = __expf(attn[i][j] - m); attn[i][j] = e; sum += e; }
        float inv = 1.f / sum;
        for (int j = 0; j < 64; ++j) attn[i][j] *= inv;
    }
    __syncthreads();

    int o = threadIdx.x & 63;
    int jb = threadIdx.x >> 6;                      // 0..3
    for (int s = 0; s < 16; ++s) {
        int j = jb + 4 * s;
        float acc = 0.f;
        for (int i = 0; i < 64; ++i)
            acc += attn[i][j] * Wproj[(h * DH + i) * 64 + o];
        Wmix[((size_t)b * HD + h * DH + j) * 64 + o] = acc;
    }
}

// ---------------------------------------------------------------------------
// K6: fold into per-batch 64x(64+64): Wall = Wv @ Wmix (attention path),
// Wfused = Wv @ Wc1d (positional path). Stored transposed bf16 [128][64].
// ---------------------------------------------------------------------------
__global__ void k_fold(const float* __restrict__ Wv,
                       const float* __restrict__ Wmix,
                       const float* __restrict__ Wc1d,
                       unsigned short* __restrict__ Wcombt) {
    int b = blockIdx.x;
    int r = threadIdx.x & 63;                       // input channel
    int ng = threadIdx.x >> 6;                      // 0..3
    const float* wm = Wmix + (size_t)b * HD * 64;
    for (int s = 0; s < 32; ++s) {
        int n = ng + 4 * s;                         // 0..127
        float acc = 0.f;
        if (n < 64) {
            for (int k = 0; k < HD; ++k) acc += Wv[r * HD + k] * wm[k * 64 + n];
        } else {
            int o = n - 64;
            for (int k = 0; k < HD; ++k) acc += Wv[r * HD + k] * Wc1d[k * 64 + o];
        }
        Wcombt[((size_t)b * 128 + n) * 64 + r] = f2bf(acc);
    }
}

// ---------------------------------------------------------------------------
// K7: full-res pass: [131072 x 64] @ [64 x 128] -> out_c (d_out) and p0 (ws)
// ---------------------------------------------------------------------------
__global__ void k_main_gemm(const unsigned short* __restrict__ x_bf,
                            const unsigned short* __restrict__ Wcombt,
                            const float* __restrict__ bproj,
                            const float* __restrict__ bc1d,
                            float* __restrict__ out,
                            float* __restrict__ p0) {
    int wave = threadIdx.x >> 5, lane = threadIdx.x & 31;
    size_t m0 = (size_t)blockIdx.x * 16;
    int b = (int)(m0 >> 16);                        // 65536 rows per batch

    const unsigned short* arow = x_bf + m0 * 64;
    v16bf a0 = load_a_frag(arow, 64);
    v16bf a1 = load_a_frag(arow + 32, 64);
    const unsigned short* brow = Wcombt + ((size_t)b * 128 + wave * 16) * 64;
    v16bf b0 = load_b_frag(brow, 64);
    v16bf b1 = load_b_frag(brow + 32, 64);

    v8f acc = {};
    acc = wmma_bf16(a0, b0, acc);
    acc = wmma_bf16(a1, b1, acc);

    int n = wave * 16 + (lane & 15);
    int hi = lane >> 4;
    if (n < 64) {
        float bias = bproj[n];
#pragma unroll
        for (int r = 0; r < 8; ++r)
            out[(m0 + hi * 8 + r) * 64 + n] = acc[r] + bias;
    } else {
        int o = n - 64;
        float bias = bc1d[o];
#pragma unroll
        for (int r = 0; r < 8; ++r)
            p0[(m0 + hi * 8 + r) * 64 + o] = acc[r] + bias;
    }
}

// ---------------------------------------------------------------------------
// K8: depthwise 3x3 (SAME, zero pad) + exact GELU
// ---------------------------------------------------------------------------
__global__ void k_dw1(const float* __restrict__ p0,
                      const float* __restrict__ Wpe1,
                      float* __restrict__ p1) {
    int idx = blockIdx.x * 256 + threadIdx.x;
    int c = idx & 63;
    int x = (idx >> 6) & 255;
    int y = (idx >> 14) & 255;
    int b = idx >> 22;
    float s = 0.f;
#pragma unroll
    for (int dy = -1; dy <= 1; ++dy) {
        int yy = y + dy; if (yy < 0 || yy > 255) continue;
#pragma unroll
        for (int dx = -1; dx <= 1; ++dx) {
            int xx = x + dx; if (xx < 0 || xx > 255) continue;
            s += p0[((((size_t)b * HH) + yy) * WW + xx) * CC + c] *
                 Wpe1[c * 9 + (dy + 1) * 3 + (dx + 1)];
        }
    }
    p1[idx] = 0.5f * s * (1.0f + erff(s * 0.70710678118654752f));
}

// ---------------------------------------------------------------------------
// K9: depthwise 3x3 + add into out
// ---------------------------------------------------------------------------
__global__ void k_dw2_add(const float* __restrict__ p1,
                          const float* __restrict__ Wpe2,
                          float* __restrict__ out) {
    int idx = blockIdx.x * 256 + threadIdx.x;
    int c = idx & 63;
    int x = (idx >> 6) & 255;
    int y = (idx >> 14) & 255;
    int b = idx >> 22;
    float s = 0.f;
#pragma unroll
    for (int dy = -1; dy <= 1; ++dy) {
        int yy = y + dy; if (yy < 0 || yy > 255) continue;
#pragma unroll
        for (int dx = -1; dx <= 1; ++dx) {
            int xx = x + dx; if (xx < 0 || xx > 255) continue;
            s += p1[((((size_t)b * HH) + yy) * WW + xx) * CC + c] *
                 Wpe2[c * 9 + (dy + 1) * 3 + (dx + 1)];
        }
    }
    out[idx] = out[idx] + s;
}

// ---------------------------------------------------------------------------
// Host launcher
// ---------------------------------------------------------------------------
extern "C" void kernel_launch(void* const* d_in, const int* in_sizes, int n_in,
                              void* d_out, int out_size, void* d_ws, size_t ws_size,
                              hipStream_t stream) {
    (void)in_sizes; (void)n_in; (void)out_size; (void)ws_size;

    const float* x_in   = (const float*)d_in[0];
    const float* Wq     = (const float*)d_in[1];
    const float* Wk     = (const float*)d_in[2];
    const float* Wv     = (const float*)d_in[3];
    const float* rescal = (const float*)d_in[4];
    const float* Wproj  = (const float*)d_in[5];
    const float* bproj  = (const float*)d_in[6];
    const float* Wc1d   = (const float*)d_in[7];
    const float* bc1d   = (const float*)d_in[8];
    const float* Wpe1   = (const float*)d_in[9];
    const float* Wpe2   = (const float*)d_in[10];
    float* out = (float*)d_out;

    // Workspace layout (bytes), all 16B-aligned.
    char* ws = (char*)d_ws;
    size_t o_xbf   = 0;                                         // 16 MiB bf16 x
    size_t o_xdbf  = o_xbf  + (size_t)BB * NTOK * CC * 2;       // 4 MiB bf16 xd
    size_t o_wqkt  = o_xdbf + (size_t)BB * NDS * CC * 2;        // 128 KiB
    size_t o_qt    = o_wqkt + (size_t)1024 * 64 * 2;            // 32 MiB Qt
    size_t o_kt    = o_qt   + (size_t)BB * HD * NDS * 2;        // 32 MiB Kt
    size_t o_invq  = o_kt   + (size_t)BB * HD * NDS * 2;
    size_t o_invk  = o_invq + (size_t)BB * HD * 4;
    size_t o_wmix  = o_invk + (size_t)BB * HD * 4;
    size_t o_wcomb = o_wmix + (size_t)BB * HD * 64 * 4;
    // p0 / p1 (each B*NTOK*64*4 = 32 MiB) alias Qt / Kt (dead after k_attn).
    unsigned short* x_bf  = (unsigned short*)(ws + o_xbf);
    unsigned short* xd_bf = (unsigned short*)(ws + o_xdbf);
    unsigned short* Wqkt  = (unsigned short*)(ws + o_wqkt);
    unsigned short* Qt    = (unsigned short*)(ws + o_qt);
    unsigned short* Kt    = (unsigned short*)(ws + o_kt);
    float* invq  = (float*)(ws + o_invq);
    float* invk  = (float*)(ws + o_invk);
    float* Wmix  = (float*)(ws + o_wmix);
    unsigned short* Wcombt = (unsigned short*)(ws + o_wcomb);
    float* p0 = (float*)(ws + o_qt);
    float* p1 = (float*)(ws + o_kt);

    k_pool_cast<<<dim3(8192), dim3(256), 0, stream>>>(x_in, x_bf, xd_bf);
    k_prep_wqk<<<dim3(256), dim3(256), 0, stream>>>(Wq, Wk, Wqkt);
    k_gemm_qk<<<dim3(2048, 8), dim3(256), 0, stream>>>(xd_bf, Wqkt, Qt, Kt);
    k_norms<<<dim3(128, 2), dim3(256), 0, stream>>>(Qt, Kt, invq, invk);
    k_attn<<<dim3(BB * HEADS), dim3(256), 0, stream>>>(Qt, Kt, invq, invk,
                                                       rescal, Wproj, Wmix);
    k_fold<<<dim3(BB), dim3(256), 0, stream>>>(Wv, Wmix, Wc1d, Wcombt);
    k_main_gemm<<<dim3(8192), dim3(256), 0, stream>>>(x_bf, Wcombt, bproj,
                                                      bc1d, out, p0);
    k_dw1<<<dim3(32768), dim3(256), 0, stream>>>(p0, Wpe1, p1);
    k_dw2_add<<<dim3(32768), dim3(256), 0, stream>>>(p1, Wpe2, out);
}
